// ETAModelMLP_15607911154252
// MI455X (gfx1250) — matrix operand order repaired and checked by reading, not compiled, not run on hardware
//
#include <hip/hip_runtime.h>
#include <hip/hip_bf16.h>

typedef __attribute__((ext_vector_type(16))) __bf16 v16bf;
typedef __attribute__((ext_vector_type(8)))  __bf16 v8bf;
typedef __attribute__((ext_vector_type(8)))  float  v8f;
typedef __attribute__((ext_vector_type(4)))  unsigned int u32x4;
typedef __attribute__((ext_vector_type(8)))  int i32x8;
typedef __attribute__((ext_vector_type(4)))  int i32x4;

#define N_NODES 8192
#define DIM     256
#define CMT     128
#define LAYERS  3
#define BATCH   1024
#define SEQLEN  20

// LDS staging for B K-slices: rows of 32 bf16 (64B) padded to 80B (TDM pad:
// 4 DWORDs every 16 DWORDs) so B-fragment reads spread across LDS banks.
#define B_ROW_ELEMS 40                    // 80 bytes pitch
#define B_BUF_ELEMS (256 * B_ROW_ELEMS)   // max NC = 256
#define B_BUF_BYTES (B_BUF_ELEMS * 2)

#if __has_include(<hip/amd_detail/amd_gfx1250_TDM.h>)
#define TDM_6ARG 1
#else
#define TDM_6ARG 0
#endif

// ---------------------------------------------------------------------------
// Issue a TDM load of the B K-slice Bt[0..NC-1][kk..kk+31] into LDS at
// ldsByteAddr, row pitch 80B. D# group0/group1 built per CDNA5 ISA 8.3/8.4.
// ---------------------------------------------------------------------------
__device__ __forceinline__ void tdm_load_b_slice(
    const __bf16* Bt, int K, int NC, unsigned ldsByteAddr, int kk)
{
    const unsigned long long ga =
        (unsigned long long)(size_t)Bt + (unsigned long long)kk * 2ull;

    u32x4 g0;
    g0[0] = 1u;                                        // count=1, user mode
    g0[1] = ldsByteAddr;                               // lds_addr
    g0[2] = (unsigned)(ga & 0xffffffffu);              // global_addr[31:0]
    g0[3] = (unsigned)((ga >> 32) & 0x1ffffffu)        // global_addr[56:32]
          | (2u << 30);                                // type = 2 ("image")

    i32x8 g1;
    g1[0] = (int)((1u << 16)                           // data_size = 2B
                | (1u << 20)                           // pad_enable
                | (3u << 22)                           // pad_interval: 16 DW
                | (3u << 25));                         // pad_amount: 4 DW
    g1[1] = (int)(((unsigned)K & 0xffffu) << 16);      // tensor_dim0[15:0]
    g1[2] = (int)(((unsigned)K >> 16)                  // tensor_dim0[31:16]
                | ((unsigned)NC << 16));               // tensor_dim1[15:0]
    g1[3] = (int)(32u << 16);                          // tile_dim0 = 32
    g1[4] = (int)(unsigned)NC;                         // tile_dim1 = NC
    g1[5] = (int)K;                                    // tensor_dim0_stride
    g1[6] = 0;
    g1[7] = 0;

    i32x4 z4 = {0, 0, 0, 0};
#if TDM_6ARG
    i32x8 z8 = {0, 0, 0, 0, 0, 0, 0, 0};
    __builtin_amdgcn_tensor_load_to_lds(g0, g1, z4, z4, z8, 0);
#else
    __builtin_amdgcn_tensor_load_to_lds(g0, g1, z4, z4, 0);
#endif
}

// ---------------------------------------------------------------------------
// bf16 WMMA GEMM: OUT[M,NC] = act( A[M,K] @ Bt[NC,K]^T + bias )
// A row-major (bf16, or f32 converted in-flight when AF32); Bt row-major.
// Per wave: 32x64 tile (2 A frags x 4 B frags = 8 WMMAs / 32-K step).
// B K-slices are TDM-DMA'd into double-buffered LDS by wave 0 and shared by
// all 8 waves of the block (s_wait_tensorcnt + workgroup barrier pipeline).
// ---------------------------------------------------------------------------
template<bool AF32>
__global__ __launch_bounds__(256) void gemm_wmma_bf16(
    const void* __restrict__ Aptr,
    const __bf16* __restrict__ Bt,
    int M, int K, int NC,
    const float* __restrict__ bias, int relu,
    __bf16* __restrict__ outN, __bf16* __restrict__ outT)
{
    __shared__ __bf16 Bsh[2][B_BUF_ELEMS];

    const int lane  = threadIdx.x & 31;
    const int half  = lane >> 4;
    const int lm    = lane & 15;
    const int wib   = threadIdx.x >> 5;     // wave in block
    const int wpb   = blockDim.x >> 5;      // 8
    const bool issuer = (wib == 0);

    const int ntm  = M >> 5;
    const int ntn  = NC >> 6;
    const int nt   = ntm * ntn;
    const int ngrp = (nt + wpb - 1) / wpb;

    const unsigned ldsBase = (unsigned)(size_t)(void*)&Bsh[0][0];

    for (int g = blockIdx.x; g < ngrp; g += gridDim.x) {
        const int tile   = g * wpb + wib;
        const bool active = tile < nt;
        const int tsafe  = active ? tile : 0;
        const int tm  = tsafe / ntn;        // tn fastest: group covers all NC cols
        const int tn  = tsafe % ntn;
        const int r0  = (tm << 5) + lm;
        const int bn0 = tn << 6;

        const float*  Af0 = nullptr; const float*  Af1 = nullptr;
        const __bf16* Ab0 = nullptr; const __bf16* Ab1 = nullptr;
        if constexpr (AF32) {
            Af0 = (const float*)Aptr + (size_t)r0 * K;
            Af1 = Af0 + (size_t)16 * K;
        } else {
            Ab0 = (const __bf16*)Aptr + (size_t)r0 * K;
            Ab1 = Ab0 + (size_t)16 * K;
        }

        v8f acc[8];
        #pragma unroll
        for (int i = 0; i < 8; ++i)
            #pragma unroll
            for (int j = 0; j < 8; ++j) acc[i][j] = 0.0f;

        // Prologue: DMA first B slice, drain TENSORcnt, publish via barrier.
        if (issuer) {
            tdm_load_b_slice(Bt, K, NC, ldsBase, 0);
            __builtin_amdgcn_s_wait_tensorcnt(0);
        }
        __syncthreads();

        int cur = 0;
        for (int kk = 0; kk < K; kk += 32) {
            // Overlap: DMA next slice into the other buffer while computing.
            if (issuer && (kk + 32) < K)
                tdm_load_b_slice(Bt, K, NC, ldsBase + (unsigned)(cur ^ 1) * B_BUF_BYTES, kk + 32);

            if (active) {
                v16bf a0, a1;
                if constexpr (AF32) {
                    v8f f00 = *(const v8f*)(Af0 + kk + half * 8);
                    v8f f01 = *(const v8f*)(Af0 + kk + 16 + half * 8);
                    v8f f10 = *(const v8f*)(Af1 + kk + half * 8);
                    v8f f11 = *(const v8f*)(Af1 + kk + 16 + half * 8);
                    #pragma unroll
                    for (int j = 0; j < 8; ++j) {
                        a0[j] = (__bf16)f00[j]; a0[8 + j] = (__bf16)f01[j];
                        a1[j] = (__bf16)f10[j]; a1[8 + j] = (__bf16)f11[j];
                    }
                    if (kk + 64 < K) {
                        __builtin_prefetch(Af0 + kk + 64, 0, 0);
                        __builtin_prefetch(Af1 + kk + 64, 0, 0);
                    }
                } else {
                    v8bf q00 = *(const v8bf*)(Ab0 + kk + half * 8);
                    v8bf q01 = *(const v8bf*)(Ab0 + kk + 16 + half * 8);
                    v8bf q10 = *(const v8bf*)(Ab1 + kk + half * 8);
                    v8bf q11 = *(const v8bf*)(Ab1 + kk + 16 + half * 8);
                    #pragma unroll
                    for (int j = 0; j < 8; ++j) {
                        a0[j] = q00[j]; a0[8 + j] = q01[j];
                        a1[j] = q10[j]; a1[8 + j] = q11[j];
                    }
                    if (kk + 64 < K) __builtin_prefetch(Ab0 + kk + 64, 0, 0);
                }
                const __bf16* bbase = &Bsh[cur][0];
                #pragma unroll
                for (int t = 0; t < 4; ++t) {
                    const __bf16* bp =
                        bbase + (size_t)(bn0 + t * 16 + lm) * B_ROW_ELEMS + half * 8;
                    v8bf l0 = *(const v8bf*)bp;
                    v8bf l1 = *(const v8bf*)(bp + 16);
                    v16bf bb;
                    #pragma unroll
                    for (int j = 0; j < 8; ++j) { bb[j] = l0[j]; bb[8 + j] = l1[j]; }
                    acc[t]     = __builtin_amdgcn_wmma_f32_16x16x32_bf16(
                                     false, a0, false, bb, (short)0, acc[t],     false, false);
                    acc[4 + t] = __builtin_amdgcn_wmma_f32_16x16x32_bf16(
                                     false, a1, false, bb, (short)0, acc[4 + t], false, false);
                }
            }

            // Next buffer ready + everyone done reading current buffer.
            if (issuer) __builtin_amdgcn_s_wait_tensorcnt(0);
            __syncthreads();
            cur ^= 1;
        }

        if (active) {
            #pragma unroll
            for (int t = 0; t < 4; ++t) {
                const int col = bn0 + t * 16 + lm;
                const float bv = bias ? bias[col] : 0.0f;
                #pragma unroll
                for (int ar = 0; ar < 2; ++ar) {
                    #pragma unroll
                    for (int j = 0; j < 8; ++j) {
                        float v = acc[ar * 4 + t][j] + bv;
                        if (relu) v = v > 0.0f ? v : 0.0f;
                        const int row = (tm << 5) + ar * 16 + half * 8 + j;
                        if (outN) outN[(size_t)row * NC + col] = (__bf16)v;
                        if (outT) outT[(size_t)col * M + row] = (__bf16)v;
                    }
                }
            }
        }
    }
}

// ---------------------------------------------------------------------------
// Softmax over cmt_weight rows -> St[C,N] bf16 (transposed) + argmax route.
// ---------------------------------------------------------------------------
__global__ __launch_bounds__(256) void softmax_route_kernel(
    const float* __restrict__ W, __bf16* __restrict__ St,
    int* __restrict__ route, int Nn, int Cc)
{
    const int wid  = blockIdx.x * (blockDim.x >> 5) + (threadIdx.x >> 5);
    const int lane = threadIdx.x & 31;
    if (wid >= Nn) return;
    const float* wrow = W + (size_t)wid * Cc;

    float v[4];
    float mx = -3.0e38f; int mi = 0;
    #pragma unroll
    for (int t = 0; t < 4; ++t) {
        const int c = lane * 4 + t;
        v[t] = wrow[c];
        if (v[t] > mx) { mx = v[t]; mi = c; }
    }
    #pragma unroll
    for (int o = 16; o > 0; o >>= 1) {
        float omx = __shfl_xor(mx, o, 32);
        int   omi = __shfl_xor(mi, o, 32);
        if (omx > mx || (omx == mx && omi < mi)) { mx = omx; mi = omi; }
    }
    float s = 0.0f;
    #pragma unroll
    for (int t = 0; t < 4; ++t) { v[t] = __expf(v[t] - mx); s += v[t]; }
    #pragma unroll
    for (int o = 16; o > 0; o >>= 1) s += __shfl_xor(s, o, 32);
    const float inv = 1.0f / s;
    #pragma unroll
    for (int t = 0; t < 4; ++t) {
        const int c = lane * 4 + t;
        St[(size_t)c * Nn + wid] = (__bf16)(v[t] * inv);
    }
    if (lane == 0) route[wid] = mi;
}

__global__ __launch_bounds__(256) void cvt_x_kernel(
    const float* __restrict__ E, __bf16* __restrict__ x, __bf16* __restrict__ xT,
    int Nn, int Dd)
{
    const int total = Nn * Dd;
    for (int idx = blockIdx.x * blockDim.x + threadIdx.x; idx < total;
         idx += gridDim.x * blockDim.x) {
        const int r = idx / Dd, d = idx - r * Dd;
        const float v = E[idx];
        x[idx] = (__bf16)v;
        xT[(size_t)d * Nn + r] = (__bf16)v;
    }
}

__global__ __launch_bounds__(256) void cvt_wT_kernel(
    const float* __restrict__ W, __bf16* __restrict__ WT, int Lk, int In, int Out)
{
    const int total = Lk * In * Out;
    for (int idx = blockIdx.x * blockDim.x + threadIdx.x; idx < total;
         idx += gridDim.x * blockDim.x) {
        const int i   = idx / (In * Out);
        const int rem = idx - i * In * Out;
        const int o   = rem / In;
        const int in_ = rem - o * In;
        WT[idx] = (__bf16)W[((size_t)i * In + in_) * Out + o];
    }
}

__global__ __launch_bounds__(256) void cvt_bf_kernel(
    const float* __restrict__ src, __bf16* __restrict__ dst, size_t total)
{
    for (size_t idx = (size_t)blockIdx.x * blockDim.x + threadIdx.x; idx < total;
         idx += (size_t)gridDim.x * blockDim.x)
        dst[idx] = (__bf16)src[idx];
}

__global__ __launch_bounds__(256) void build_xcat_kernel(
    const __bf16* __restrict__ x, const __bf16* __restrict__ cmt_out,
    const int* __restrict__ route, __bf16* __restrict__ xcat, int Nn, int Dd)
{
    const int total = Nn * 2 * Dd;
    for (int idx = blockIdx.x * blockDim.x + threadIdx.x; idx < total;
         idx += gridDim.x * blockDim.x) {
        const int r = idx / (2 * Dd);
        const int j = idx - r * 2 * Dd;
        xcat[idx] = (j < Dd) ? x[(size_t)r * Dd + j]
                             : cmt_out[(size_t)route[r] * Dd + (j - Dd)];
    }
}

__global__ __launch_bounds__(256) void eta_head_kernel(
    const int* __restrict__ token, const __bf16* __restrict__ x,
    const float* __restrict__ Wout, const float* __restrict__ bout,
    float* __restrict__ out, int Bb, int Sq, int Dd)
{
    const int wid  = blockIdx.x * (blockDim.x >> 5) + (threadIdx.x >> 5);
    const int lane = threadIdx.x & 31;
    if (wid >= Bb) return;
    const int t0 = token[(size_t)wid * Sq];
    const int t1 = token[(size_t)wid * Sq + Sq - 1];
    float acc = 0.0f;
    for (int d = lane; d < Dd; d += 32)
        acc += (float)x[(size_t)t0 * Dd + d] * Wout[d]
             + (float)x[(size_t)t1 * Dd + d] * Wout[Dd + d];
    #pragma unroll
    for (int o = 16; o > 0; o >>= 1) acc += __shfl_xor(acc, o, 32);
    if (lane == 0) out[wid] = acc + bout[0];
}

// ---------------------------------------------------------------------------
static inline int gemm_blocks(int M, int NC) {
    int nt   = (M >> 5) * (NC >> 6);
    int ngrp = (nt + 7) / 8;
    if (ngrp < 1) ngrp = 1;
    if (ngrp > 1024) ngrp = 1024;
    return ngrp;
}

extern "C" void kernel_launch(void* const* d_in, const int* in_sizes, int n_in,
                              void* d_out, int out_size, void* d_ws, size_t ws_size,
                              hipStream_t stream) {
    (void)in_sizes; (void)n_in; (void)out_size;
    const int*   token = (const int*)d_in[0];
    const float* adj   = (const float*)d_in[1];
    const float* emb   = (const float*)d_in[2];
    const float* cmtw  = (const float*)d_in[3];
    const float* Wn    = (const float*)d_in[4];
    const float* bn    = (const float*)d_in[5];
    const float* Wc    = (const float*)d_in[6];
    const float* bc    = (const float*)d_in[7];
    const float* Wout  = (const float*)d_in[8];
    const float* bout  = (const float*)d_in[9];
    float* out = (float*)d_out;

    size_t off = 0;
    auto carve = [&](size_t bytes) -> void* {
        void* p = (char*)d_ws + off;
        off += (bytes + 255) & ~(size_t)255;
        return p;
    };
    __bf16* St     = (__bf16*)carve((size_t)CMT * N_NODES * 2);
    __bf16* ASt    = (__bf16*)carve((size_t)CMT * N_NODES * 2);
    __bf16* cmtAdj = (__bf16*)carve((size_t)CMT * CMT * 2);
    __bf16* xbf    = (__bf16*)carve((size_t)N_NODES * DIM * 2);
    __bf16* xTbf   = (__bf16*)carve((size_t)DIM * N_NODES * 2);
    __bf16* cmtEmb = (__bf16*)carve((size_t)CMT * DIM * 2);
    __bf16* t1T    = (__bf16*)carve((size_t)DIM * CMT * 2);
    __bf16* cmtOut = (__bf16*)carve((size_t)CMT * DIM * 2);
    __bf16* xcat   = (__bf16*)carve((size_t)N_NODES * 2 * DIM * 2);
    __bf16* yT     = (__bf16*)carve((size_t)DIM * N_NODES * 2);
    __bf16* WcT    = (__bf16*)carve((size_t)LAYERS * DIM * DIM * 2);
    __bf16* WnT    = (__bf16*)carve((size_t)LAYERS * DIM * 2 * DIM * 2);
    int*    route  = (int*)carve((size_t)N_NODES * 4);

    // bf16 adj copy when workspace allows (128 MB -> L2-resident re-reads).
    const size_t adjBytes = (size_t)N_NODES * N_NODES * 2;
    __bf16* adjBf = nullptr;
    if (ws_size >= off + adjBytes + 256) adjBf = (__bf16*)carve(adjBytes);

    softmax_route_kernel<<<N_NODES / 8, 256, 0, stream>>>(cmtw, St, route, N_NODES, CMT);
    cvt_x_kernel<<<2048, 256, 0, stream>>>(emb, xbf, xTbf, N_NODES, DIM);
    cvt_wT_kernel<<<768,  256, 0, stream>>>(Wc, WcT, LAYERS, DIM, DIM);
    cvt_wT_kernel<<<1536, 256, 0, stream>>>(Wn, WnT, LAYERS, 2 * DIM, DIM);
    if (adjBf)
        cvt_bf_kernel<<<8192, 256, 0, stream>>>(adj, adjBf, (size_t)N_NODES * N_NODES);

    // G1: ASt = (adj @ S)^T
    if (adjBf)
        gemm_wmma_bf16<false><<<gemm_blocks(N_NODES, CMT), 256, 0, stream>>>(
            adjBf, St, N_NODES, N_NODES, CMT, nullptr, 0, nullptr, ASt);
    else
        gemm_wmma_bf16<true><<<gemm_blocks(N_NODES, CMT), 256, 0, stream>>>(
            adj, St, N_NODES, N_NODES, CMT, nullptr, 0, nullptr, ASt);

    // G2: cmt_adj = S^T @ AS
    gemm_wmma_bf16<false><<<gemm_blocks(CMT, CMT), 256, 0, stream>>>(
        St, ASt, CMT, N_NODES, CMT, nullptr, 0, cmtAdj, nullptr);

    for (int i = 0; i < LAYERS; ++i) {
        gemm_wmma_bf16<false><<<gemm_blocks(CMT, DIM), 256, 0, stream>>>(
            St, xTbf, CMT, N_NODES, DIM, nullptr, 0, cmtEmb, nullptr);
        gemm_wmma_bf16<false><<<gemm_blocks(CMT, DIM), 256, 0, stream>>>(
            cmtEmb, WcT + (size_t)i * DIM * DIM, CMT, DIM, DIM, nullptr, 0, nullptr, t1T);
        gemm_wmma_bf16<false><<<gemm_blocks(CMT, DIM), 256, 0, stream>>>(
            cmtAdj, t1T, CMT, CMT, DIM, bc + (size_t)i * DIM, 1, cmtOut, nullptr);
        build_xcat_kernel<<<4096, 256, 0, stream>>>(xbf, cmtOut, route, xcat, N_NODES, DIM);
        gemm_wmma_bf16<false><<<gemm_blocks(N_NODES, DIM), 256, 0, stream>>>(
            xcat, WnT + (size_t)i * DIM * 2 * DIM, N_NODES, 2 * DIM, DIM,
            nullptr, 0, nullptr, yT);
        if (adjBf)
            gemm_wmma_bf16<false><<<gemm_blocks(N_NODES, DIM), 256, 0, stream>>>(
                adjBf, yT, N_NODES, N_NODES, DIM, bn + (size_t)i * DIM, 1, xbf, xTbf);
        else
            gemm_wmma_bf16<true><<<gemm_blocks(N_NODES, DIM), 256, 0, stream>>>(
                adj, yT, N_NODES, N_NODES, DIM, bn + (size_t)i * DIM, 1, xbf, xTbf);
    }

    eta_head_kernel<<<BATCH / 8, 256, 0, stream>>>(
        token, xbf, Wout, bout, out, BATCH, SEQLEN, DIM);
}
